// CRN_env_52286931861853
// MI455X (gfx1250) — matrix operand order repaired and checked by reading, not compiled, not run on hardware
//
#include <hip/hip_runtime.h>
#include <hip/hip_bf16.h>

typedef __attribute__((ext_vector_type(16))) __bf16 v16bf;
typedef __attribute__((ext_vector_type(8)))  __bf16 v8bf;
typedef __attribute__((ext_vector_type(8)))  float  v8f;

#define B_SZ   128
#define T_SZ   1024
#define NSER   35
#define Y_SZ   8
#define D_SZ   43
#define H_SZ   256
#define G_SZ   1024   // 4*H
#define NCONT  50
#define NBIN   10
#define K_IH   64     // D padded to WMMA K multiple
#define K_PRED 288    // (H + Y) padded to 9*32

__device__ __forceinline__ float sigmoid_f(float v) {
  return 1.0f / (1.0f + __expf(-v));
}

// gfx1250 hardware tanh (TRANS32). One independent op / V_NOP required after a
// TRANS op before its result is consumed (ISA 7.4), so pad with v_nop.
__device__ __forceinline__ float tanh_hw(float x) {
  float r;
  asm volatile("v_tanh_f32 %0, %1\n\tv_nop" : "=v"(r) : "v"(x));
  return r;
}

__device__ __forceinline__ v16bf join16(v8bf lo, v8bf hi) {
  return __builtin_shufflevector(lo, hi, 0, 1, 2, 3, 4, 5, 6, 7,
                                 8, 9, 10, 11, 12, 13, 14, 15);
}

// A fragment (16x32 bf16, MxK) from row-major [16][ld] LDS tile.
// lane (l16, hf): element e -> K = kb + hf*16 + e
__device__ __forceinline__ v16bf frag_a(const __bf16* base, int ld, int l16,
                                        int hf, int kb) {
  const __bf16* p0 = base + l16 * ld + kb + hf * 16;
  v8bf lo = *(const v8bf*)p0;        // 16B aligned by layout construction
  v8bf hi = *(const v8bf*)(p0 + 8);
  return join16(lo, hi);
}

// B fragment (32x16 bf16, KxN). row_p = &W[n=l16][hf*16]; fragment is the
// 32 contiguous bytes at row_p + kb:  element e -> K = kb + 16*hf + e
__device__ __forceinline__ v16bf frag_b_row(const __bf16* row_p, int kb) {
  v8bf lo = *(const v8bf*)(row_p + kb);
  v8bf hi = *(const v8bf*)(row_p + kb + 8);
  return join16(lo, hi);
}

__device__ __forceinline__ v8f wmma_bf16(v16bf a, v16bf b, v8f c) {
  return __builtin_amdgcn_wmma_f32_16x16x32_bf16(false, a, false, b, (short)0, c,
                                                 false, false);
}

// ---------------- weight prep: fp32 -> bf16 (optionally zero-padded) -------
__global__ void k_cvt_plain(const float* __restrict__ src, __bf16* __restrict__ dst, int n) {
  int i = blockIdx.x * 256 + threadIdx.x;
  if (i < n) dst[i] = (__bf16)src[i];
}

__global__ void k_cvt_pad(const float* __restrict__ src, __bf16* __restrict__ dst,
                          int N, int Nsrc, int Ksrc, int Kdst) {
  int i = blockIdx.x * 256 + threadIdx.x;
  if (i >= N * Kdst) return;
  int n = i / Kdst, k = i - n * Kdst;
  float v = (n < Nsrc && k < Ksrc) ? src[n * Ksrc + k] : 0.0f;
  dst[i] = (__bf16)v;
}

// ---------------- LSTM scan: 8 blocks x 16 batch rows, 16 waves/block ------
__global__ __launch_bounds__(512, 1) void k_lstm_scan(
    const float* __restrict__ x, const float* __restrict__ b_ih,
    const float* __restrict__ b_hh, const __bf16* __restrict__ Whh,
    const __bf16* __restrict__ Wih, __bf16* __restrict__ out_bf) {
  __shared__ float  gates[16][G_SZ];   // 64 KB
  __shared__ __bf16 h_bf[16][H_SZ];    //  8 KB
  __shared__ __bf16 x_bf[16][K_IH];    //  2 KB
  __shared__ float  bias_s[G_SZ];      //  4 KB

  const int tid  = threadIdx.x;
  const int wave = tid >> 5;          // 0..15, owns gate cols [wave*64, wave*64+64)
  const int lane = tid & 31;
  const int l16  = lane & 15;
  const int hf   = lane >> 4;
  const int bm   = blockIdx.x * 16;   // batch base

  // fixed elementwise slice for this thread: row em, cols en0..en0+7
  const int em  = tid >> 5;
  const int en0 = (tid & 31) * 8;
  float c_r[8];                       // cell state lives in registers
#pragma unroll
  for (int j = 0; j < 8; ++j) c_r[j] = 0.0f;

  for (int i = tid; i < G_SZ; i += 512) bias_s[i] = b_ih[i] + b_hh[i];
  for (int i = tid; i < 16 * H_SZ; i += 512) h_bf[i >> 8][i & 255] = (__bf16)0.0f;
  __syncthreads();

  for (int t = 0; t < T_SZ; ++t) {
    // stage xin_t tile (16 x 64): [series(35) | prev_y(8) | pad]
    for (int i = tid; i < 16 * K_IH; i += 512) {
      int m = i >> 6, k = i & 63;
      float v = 0.0f;
      if (k < NSER) v = x[((size_t)(bm + m) * T_SZ + t) * D_SZ + k];
      else if (k < D_SZ && t > 0) v = x[((size_t)(bm + m) * T_SZ + (t - 1)) * D_SZ + k];
      x_bf[m][k] = (__bf16)v;
    }
    if (t + 1 < T_SZ && tid < 16)
      __builtin_prefetch(&x[((size_t)(bm + tid) * T_SZ + (t + 1)) * D_SZ], 0, 1);
    __syncthreads();

    // preload A fragments once per step; reused by all 4 n-tiles
    v16bf ax0 = frag_a(&x_bf[0][0], K_IH, l16, hf, 0);
    v16bf ax1 = frag_a(&x_bf[0][0], K_IH, l16, hf, 32);
    v16bf ah[8];
#pragma unroll
    for (int kt = 0; kt < 8; ++kt)
      ah[kt] = frag_a(&h_bf[0][0], H_SZ, l16, hf, kt * 32);

    // gates = bias + xin @ Wih^T + h @ Whh^T  (bf16 WMMA, fp32 acc).
    // nt loop NOT unrolled: keeps weight-fragment loads non-invariant so they
    // cannot be hoisted out of the t-loop (avoids VGPR spills).
#pragma clang loop unroll(disable)
    for (int nt = 0; nt < 4; ++nt) {
      const int ncol = wave * 64 + nt * 16;
      const __bf16* wih_p = Wih + (size_t)(ncol + l16) * K_IH + hf * 16;
      const __bf16* whh_p = Whh + (size_t)(ncol + l16) * H_SZ + hf * 16;
      v8f acc = {};
      acc = wmma_bf16(ax0, frag_b_row(wih_p, 0), acc);
      acc = wmma_bf16(ax1, frag_b_row(wih_p, 32), acc);
#pragma unroll
      for (int kt = 0; kt < 8; ++kt)
        acc = wmma_bf16(ah[kt], frag_b_row(whh_p, kt * 32), acc);
      const float bb = bias_s[ncol + l16];
#pragma unroll
      for (int r = 0; r < 8; ++r)
        gates[r + hf * 8][ncol + l16] = acc[r] + bb;
    }
    __syncthreads();

    // elementwise cell update on this thread's fixed (em, en0..en0+7) slice;
    // h written as one 16B vector store to LDS and to global
    v8bf hv;
#pragma unroll
    for (int j = 0; j < 8; ++j) {
      int n = en0 + j;
      float ig = sigmoid_f(gates[em][n]);
      float fg = sigmoid_f(gates[em][n + 256]);
      float gg = tanh_hw(gates[em][n + 512]);
      float og = sigmoid_f(gates[em][n + 768]);
      float c  = fg * c_r[j] + ig * gg;
      c_r[j]   = c;
      hv[j]    = (__bf16)(og * tanh_hw(c));
    }
    *(v8bf*)&h_bf[em][en0] = hv;
    *(v8bf*)&out_bf[((size_t)(bm + em) * T_SZ + t) * H_SZ + en0] = hv;
    __syncthreads();
  }
}

// ---------------- heads: one 16-row M-tile of flattened (b,t) per block ----
__global__ __launch_bounds__(256, 2) void k_heads(
    const float* __restrict__ x, const __bf16* __restrict__ out_bf,
    const __bf16* __restrict__ Wafc1, const __bf16* __restrict__ Waout,
    const __bf16* __restrict__ Wcont, const __bf16* __restrict__ Wbin,
    const float* __restrict__ b_afc1, const float* __restrict__ b_aout,
    const float* __restrict__ b_cont, const float* __restrict__ b_bin,
    float* __restrict__ cont_out, float* __restrict__ bin_out,
    float* __restrict__ act_out) {
  __shared__ __bf16 A1[16][K_PRED];   // pred = [out(256) | cur_y(8) | pad]
  __shared__ __bf16 Hh[16][H_SZ];     // relu(out@Wafc1^T + b) in bf16

  const int tid  = threadIdx.x;
  const int wave = tid >> 5;          // 0..7
  const int lane = tid & 31;
  const int l16  = lane & 15;
  const int hf   = lane >> 4;
  const size_t row0 = (size_t)blockIdx.x * 16;   // flattened b*T + t

  for (int i = tid; i < 16 * K_PRED; i += 256) {
    int m = i / K_PRED, k = i - m * K_PRED;
    size_t row = row0 + m;
    float v = 0.0f;
    if (k < H_SZ) v = (float)out_bf[row * H_SZ + k];
    else if (k < H_SZ + Y_SZ) v = x[row * D_SZ + NSER + (k - H_SZ)];
    A1[m][k] = (__bf16)v;
  }
  __syncthreads();

  // afc1: 16 N-tiles over 8 waves (2 each), K = 256
#pragma clang loop unroll(disable)
  for (int j = 0; j < 2; ++j) {
    const int ncol = (wave * 2 + j) * 16;
    const __bf16* w_p = Wafc1 + (size_t)(ncol + l16) * H_SZ + hf * 16;
    v8f acc = {};
#pragma unroll 4
    for (int kt = 0; kt < 8; ++kt) {
      v16bf a = frag_a(&A1[0][0], K_PRED, l16, hf, kt * 32);
      v16bf b = frag_b_row(w_p, kt * 32);
      acc = wmma_bf16(a, b, acc);
    }
#pragma unroll
    for (int r = 0; r < 8; ++r) {
      int m = r + hf * 8, n = ncol + l16;
      float v = acc[r] + b_afc1[n];
      Hh[m][n] = (__bf16)(v > 0.0f ? v : 0.0f);
    }
  }
  __syncthreads();

  if (wave == 0) {                       // aout: N padded 8->16, K=256
    const __bf16* w_p = Waout + (size_t)l16 * H_SZ + hf * 16;
    v8f acc = {};
#pragma unroll 4
    for (int kt = 0; kt < 8; ++kt) {
      v16bf a = frag_a(&Hh[0][0], H_SZ, l16, hf, kt * 32);
      v16bf b = frag_b_row(w_p, kt * 32);
      acc = wmma_bf16(a, b, acc);
    }
#pragma unroll
    for (int r = 0; r < 8; ++r) {
      int m = r + hf * 8;
      if (l16 < Y_SZ) act_out[(row0 + m) * Y_SZ + l16] = acc[r] + b_aout[l16];
    }
  } else if (wave <= 4) {                // cont: N padded 50->64, K=288
    const int ncol = (wave - 1) * 16;
    const __bf16* w_p = Wcont + (size_t)(ncol + l16) * K_PRED + hf * 16;
    v8f acc = {};
#pragma unroll 3
    for (int kt = 0; kt < 9; ++kt) {
      v16bf a = frag_a(&A1[0][0], K_PRED, l16, hf, kt * 32);
      v16bf b = frag_b_row(w_p, kt * 32);
      acc = wmma_bf16(a, b, acc);
    }
#pragma unroll
    for (int r = 0; r < 8; ++r) {
      int m = r + hf * 8, n = ncol + l16;
      if (n < NCONT) cont_out[(row0 + m) * NCONT + n] = acc[r] + b_cont[n];
    }
  } else if (wave == 5) {                // bin: N padded 10->16, K=288
    const __bf16* w_p = Wbin + (size_t)l16 * K_PRED + hf * 16;
    v8f acc = {};
#pragma unroll 3
    for (int kt = 0; kt < 9; ++kt) {
      v16bf a = frag_a(&A1[0][0], K_PRED, l16, hf, kt * 32);
      v16bf b = frag_b_row(w_p, kt * 32);
      acc = wmma_bf16(a, b, acc);
    }
#pragma unroll
    for (int r = 0; r < 8; ++r) {
      int m = r + hf * 8;
      if (l16 < NBIN)
        bin_out[(row0 + m) * NBIN + l16] = sigmoid_f(acc[r] + b_bin[l16]);
    }
  }
}

// ---------------- workspace layout (bytes) ---------------------------------
constexpr size_t OFF_WHH   = 0;
constexpr size_t OFF_WIH   = OFF_WHH  + (size_t)G_SZ * H_SZ * 2;   // 512K
constexpr size_t OFF_AFC1  = OFF_WIH  + (size_t)G_SZ * K_IH * 2;   // +128K
constexpr size_t OFF_AOUT  = OFF_AFC1 + (size_t)H_SZ * H_SZ * 2;   // +128K
constexpr size_t OFF_CONT  = OFF_AOUT + (size_t)16 * H_SZ * 2;
constexpr size_t OFF_BIN   = OFF_CONT + (size_t)64 * K_PRED * 2;
constexpr size_t OFF_OUTBF = 841728;                               // 256-aligned
// out_bf: B*T*H bf16 = 64 MB

extern "C" void kernel_launch(void* const* d_in, const int* in_sizes, int n_in,
                              void* d_out, int out_size, void* d_ws, size_t ws_size,
                              hipStream_t stream) {
  const float* x      = (const float*)d_in[0];
  const float* W_ih   = (const float*)d_in[1];
  const float* W_hh   = (const float*)d_in[2];
  const float* b_ih   = (const float*)d_in[3];
  const float* b_hh   = (const float*)d_in[4];
  const float* W_afc1 = (const float*)d_in[5];
  const float* b_afc1 = (const float*)d_in[6];
  const float* W_aout = (const float*)d_in[7];
  const float* b_aout = (const float*)d_in[8];
  const float* W_cont = (const float*)d_in[9];
  const float* b_cont = (const float*)d_in[10];
  const float* W_bin  = (const float*)d_in[11];
  const float* b_bin  = (const float*)d_in[12];

  char* ws = (char*)d_ws;
  __bf16* Whh_bf  = (__bf16*)(ws + OFF_WHH);
  __bf16* Wih_bf  = (__bf16*)(ws + OFF_WIH);
  __bf16* Wafc1_b = (__bf16*)(ws + OFF_AFC1);
  __bf16* Waout_b = (__bf16*)(ws + OFF_AOUT);
  __bf16* Wcont_b = (__bf16*)(ws + OFF_CONT);
  __bf16* Wbin_b  = (__bf16*)(ws + OFF_BIN);
  __bf16* out_bf  = (__bf16*)(ws + OFF_OUTBF);

  float* cont_out = (float*)d_out;
  float* bin_out  = cont_out + (size_t)B_SZ * T_SZ * NCONT;
  float* act_out  = bin_out  + (size_t)B_SZ * T_SZ * NBIN;

  // weight conversion / padding
  k_cvt_plain<<<(G_SZ * H_SZ + 255) / 256, 256, 0, stream>>>(W_hh, Whh_bf, G_SZ * H_SZ);
  k_cvt_pad  <<<(G_SZ * K_IH + 255) / 256, 256, 0, stream>>>(W_ih, Wih_bf, G_SZ, G_SZ, D_SZ, K_IH);
  k_cvt_plain<<<(H_SZ * H_SZ + 255) / 256, 256, 0, stream>>>(W_afc1, Wafc1_b, H_SZ * H_SZ);
  k_cvt_pad  <<<(16 * H_SZ + 255) / 256, 256, 0, stream>>>(W_aout, Waout_b, 16, Y_SZ, H_SZ, H_SZ);
  k_cvt_pad  <<<(64 * K_PRED + 255) / 256, 256, 0, stream>>>(W_cont, Wcont_b, 64, NCONT, H_SZ + Y_SZ, K_PRED);
  k_cvt_pad  <<<(16 * K_PRED + 255) / 256, 256, 0, stream>>>(W_bin, Wbin_b, 16, NBIN, H_SZ + Y_SZ, K_PRED);

  // sequential LSTM scan, partitioned over batch (no inter-block sync needed)
  k_lstm_scan<<<B_SZ / 16, 512, 0, stream>>>(x, b_ih, b_hh, Whh_bf, Wih_bf, out_bf);

  // per-(b,t) head GEMMs
  k_heads<<<(B_SZ * T_SZ) / 16, 256, 0, stream>>>(
      x, out_bf, Wafc1_b, Waout_b, Wcont_b, Wbin_b,
      b_afc1, b_aout, b_cont, b_bin, cont_out, bin_out, act_out);
}
